// Attention_60241211293711
// MI455X (gfx1250) — matrix operand order repaired and checked by reading, not compiled, not run on hardware
//
#include <hip/hip_runtime.h>
#include <hip/hip_bf16.h>

typedef __attribute__((ext_vector_type(16))) __bf16 v16bf;
typedef __attribute__((ext_vector_type(8)))  float  v8f;

union Frag {
    v16bf v;
    uint4 q[2];
};

__device__ __forceinline__ unsigned short f2bf(float f) {
    union { float f; unsigned u; } x;
    x.f = f;
    unsigned r = x.u + 0x7FFFu + ((x.u >> 16) & 1u);   // round-to-nearest-even
    return (unsigned short)(r >> 16);
}

// ---------------------------------------------------------------------------
// fp32 -> bf16 elementwise conversion
// ---------------------------------------------------------------------------
__global__ void f32_to_bf16_kernel(const float* __restrict__ in,
                                   unsigned short* __restrict__ out, int n) {
    for (int i = blockIdx.x * blockDim.x + threadIdx.x; i < n;
         i += gridDim.x * blockDim.x)
        out[i] = f2bf(in[i]);
}

// ---------------------------------------------------------------------------
// Precompute padded relative-position bias: bias_pad[12][208][224] fp32.
// Key padding (tk >= 196) is baked in as -1e30 (softmax mask).
// ---------------------------------------------------------------------------
__global__ void build_bias_kernel(const float* __restrict__ rpb,
                                  float* __restrict__ bias_pad) {
    const int i = blockIdx.x * blockDim.x + threadIdx.x;
    const int total = 12 * 208 * 224;
    if (i >= total) return;
    const int h = i / (208 * 224);
    const int rem = i - h * 208 * 224;
    const int tq = rem / 224;
    const int tk = rem - tq * 224;
    float v;
    if (tk >= 196) {
        v = -1e30f;
    } else {
        const int tqc = tq < 196 ? tq : 195;
        const int qi = tqc / 14, qj = tqc - qi * 14;
        const int ki = tk / 14, kj = tk - ki * 14;
        v = rpb[((qi - ki + 13) * 27 + (qj - kj + 13)) * 12 + h];
    }
    bias_pad[i] = v;
}

// ---------------------------------------------------------------------------
// Async LDS-DMA staging: copy 64B (4 x b128) global -> LDS, VGPR-bypassing.
// GVS mode: SADDR (uniform 64-bit base) + 32-bit VADDR offset; INST_OFFSET is
// added to both the global and LDS addresses (ISA 08_async_tensor §4.4).
// ---------------------------------------------------------------------------
__device__ __forceinline__ void async_copy64B(unsigned lds_off, unsigned gvoff,
                                              const void* sbase) {
    asm volatile(
        "global_load_async_to_lds_b128 %0, %1, %2 offset:0\n\t"
        "global_load_async_to_lds_b128 %0, %1, %2 offset:16\n\t"
        "global_load_async_to_lds_b128 %0, %1, %2 offset:32\n\t"
        "global_load_async_to_lds_b128 %0, %1, %2 offset:48"
        :: "v"(lds_off), "v"(gvoff), "s"(sbase) : "memory");
}

__device__ __forceinline__ void wait_async0() {
    asm volatile("s_wait_asynccnt 0x0" ::: "memory");
}

// ---------------------------------------------------------------------------
// Tiled bf16 WMMA GEMM:  out[m][n] = sum_k A[m][k] * W[n][k] (+ bias)
// MODE 0: qkv bias split [q_bias, 0, v_bias]; q rows pre-scaled by 0.125;
//         bf16 output.
// MODE 1: proj bias, fp32 output.
// BM=BN=128, BK=32, 256 threads = 8 waves in 2(M) x 4(N) layout,
// each wave computes 64x32 = 4x2 WMMA accumulators.
// ---------------------------------------------------------------------------
template <int MODE>
__global__ __launch_bounds__(256)
void gemm_bf16_kernel(const unsigned short* __restrict__ A,
                      const unsigned short* __restrict__ W,
                      unsigned short* __restrict__ outb,
                      float* __restrict__ outf,
                      const float* __restrict__ qb,
                      const float* __restrict__ vb,
                      const float* __restrict__ pb,
                      int M, int N, int K) {
    constexpr int BM = 128, BN = 128, BK = 32, LDT = 40;  // 40 ush = 80B row pitch
    __shared__ unsigned short As[BM * LDT];
    __shared__ unsigned short Bs[BN * LDT];

    const int tid  = threadIdx.x;
    const int lane = tid & 31;
    const int wid  = tid >> 5;
    const int wm   = wid & 1;   // 0..1
    const int wn   = wid >> 1;  // 0..3
    const int l15  = lane & 15;
    const int hi   = lane >> 4; // 0/1
    const int m0   = blockIdx.y * BM;
    const int n0   = blockIdx.x * BN;

    // staging setup (wave-uniform branch: tid<128 covers waves 0-3 exactly)
    const int r = tid & 127;
    const unsigned short* sbase =
        (tid < 128) ? A + (size_t)m0 * K : W + (size_t)n0 * K;
    const unsigned lds_off =
        (unsigned)(uintptr_t)((tid < 128) ? &As[r * LDT] : &Bs[r * LDT]);
    const unsigned row_byte = (unsigned)r * (unsigned)K * 2u;

    v8f acc[4][2];
#pragma unroll
    for (int i = 0; i < 4; i++)
#pragma unroll
        for (int j = 0; j < 2; j++)
            acc[i][j] = (v8f){0.f, 0.f, 0.f, 0.f, 0.f, 0.f, 0.f, 0.f};

    for (int k0 = 0; k0 < K; k0 += BK) {
        __syncthreads();                       // prior iteration's readers done
        async_copy64B(lds_off, row_byte + (unsigned)k0 * 2u, sbase);
        if (k0 + BK < K)
            __builtin_prefetch((const char*)sbase + row_byte + (k0 + BK) * 2, 0, 0);
        wait_async0();                         // our wave's DMA complete
        __syncthreads();                       // all waves' tiles visible

        // --- fragments from LDS
        Frag af[4], bfr[2];
#pragma unroll
        for (int i = 0; i < 4; i++) {
            const unsigned short* p = &As[(wm * 64 + i * 16 + l15) * LDT];
            af[i].q[0] = *(const uint4*)(p + hi * 16);       // K hi*16 .. +7
            af[i].q[1] = *(const uint4*)(p + hi * 16 + 8);   // K hi*16+8 .. +15
        }
#pragma unroll
        for (int j = 0; j < 2; j++) {
            const unsigned short* p = &Bs[(wn * 32 + j * 16 + l15) * LDT] + hi * 16;
            bfr[j].q[0] = ((const uint4*)p)[0];              // K hi*16 .. +15
            bfr[j].q[1] = ((const uint4*)p)[1];
        }
#pragma unroll
        for (int i = 0; i < 4; i++)
#pragma unroll
            for (int j = 0; j < 2; j++)
                acc[i][j] = __builtin_amdgcn_wmma_f32_16x16x32_bf16(
                    false, af[i].v, false, bfr[j].v, (short)0, acc[i][j],
                    false, false);
    }

    // --- epilogue (C/D layout: VGPR v -> row v + hi*8, lane -> col l15)
#pragma unroll
    for (int i = 0; i < 4; i++)
#pragma unroll
        for (int j = 0; j < 2; j++) {
            const int mbase = m0 + wm * 64 + i * 16 + hi * 8;
            const int n = n0 + wn * 32 + j * 16 + l15;
#pragma unroll
            for (int v = 0; v < 8; v++) {
                const int m = mbase + v;
                float val = acc[i][j][v];
                if (MODE == 0) {
                    const int s = n / 768;
                    const int rr = n - s * 768;
                    if (s == 0)      val = (val + qb[rr]) * 0.125f;  // fold scale
                    else if (s == 2) val = val + vb[rr];
                    outb[(size_t)m * N + n] = f2bf(val);
                } else {
                    outf[(size_t)m * N + n] = val + pb[n];
                }
            }
        }
}

// ---------------------------------------------------------------------------
// Windowed attention, one workgroup per (b, h).
// qkvb: [25088][2304] bf16 (cols: [q|k|v] x [12 heads] x [64]); q pre-scaled.
// bias_pad: [12][208][224] fp32 (mask baked in).
// ab  : [25088][768]  bf16 attention output (pre-projection)
// ---------------------------------------------------------------------------
__global__ __launch_bounds__(256)
void attn_kernel(const unsigned short* __restrict__ qkvb,
                 const float* __restrict__ bias_pad,
                 unsigned short* __restrict__ ab) {
    constexpr int LVT = 240;                       // VT row pitch (ush)
    __shared__ unsigned short VT[64 * LVT];        // V transposed [d][t2], 30 KB
    __shared__ unsigned short Pl[8][16 * 224];     // per-wave P staging, 56 KB

    const int bh = blockIdx.x;
    const int b = bh / 12, h = bh % 12;
    const int tid = threadIdx.x;
    const int lane = tid & 31, wid = tid >> 5;
    const int l15 = lane & 15, hi = lane >> 4;
    const size_t base = (size_t)b * 196 * 2304;

    // ---- stage V transposed into LDS (8 bf16 per chunk, scattered by d)
    for (int c = tid; c < 196 * 8; c += 256) {
        const int t2 = c >> 3, db = (c & 7) * 8;
        uint4 x = *(const uint4*)(qkvb + base + (size_t)t2 * 2304 + 1536 + h * 64 + db);
        const unsigned short* u = (const unsigned short*)&x;
#pragma unroll
        for (int j = 0; j < 8; j++) VT[(db + j) * LVT + t2] = u[j];
    }
    for (int c = tid; c < 64 * 28; c += 256) {     // zero the key padding 196..223
        const int d = c / 28, t2 = 196 + c % 28;
        VT[d * LVT + t2] = 0;
    }
    __syncthreads();

    for (int mt = wid; mt < 13; mt += 8) {         // 13 query tiles of 16
        // ---- Q fragments (A layout) straight from global
        Frag qf[2];
        const int tq = mt * 16 + l15;
        const int tqc = tq < 196 ? tq : 195;
        const unsigned short* qp = qkvb + base + (size_t)tqc * 2304 + h * 64;
#pragma unroll
        for (int kk = 0; kk < 2; kk++) {
            qf[kk].q[0] = *(const uint4*)(qp + kk * 32 + hi * 16);
            qf[kk].q[1] = *(const uint4*)(qp + kk * 32 + hi * 16 + 8);
        }

        // ---- S = Q K^T over 14 key tiles (224 padded keys)
        v8f acc[14];
#pragma unroll
        for (int nt = 0; nt < 14; nt++)
            acc[nt] = (v8f){0.f, 0.f, 0.f, 0.f, 0.f, 0.f, 0.f, 0.f};
#pragma unroll
        for (int nt = 0; nt < 14; nt++) {
            const int tk = nt * 16 + l15;
            const int tkc = tk < 196 ? tk : 195;
            const unsigned short* kp =
                qkvb + base + (size_t)tkc * 2304 + 768 + h * 64 + hi * 16;
#pragma unroll
            for (int kk = 0; kk < 2; kk++) {
                Frag kf;
                kf.q[0] = ((const uint4*)(kp + kk * 32))[0];
                kf.q[1] = ((const uint4*)(kp + kk * 32))[1];
                acc[nt] = __builtin_amdgcn_wmma_f32_16x16x32_bf16(
                    false, qf[kk].v, false, kf.v, (short)0, acc[nt], false, false);
            }
        }

        // ---- add precomputed bias (mask baked in); base + immediate offsets
        {
            const float* bp =
                bias_pad + ((size_t)h * 208 + mt * 16 + hi * 8) * 224 + l15;
#pragma unroll
            for (int nt = 0; nt < 14; nt++)
#pragma unroll
                for (int v = 0; v < 8; v++)
                    acc[nt][v] += bp[v * 224 + nt * 16];
        }

        // ---- softmax: rows live on 16 lanes (lanes 0-15: rows hi*8+v etc.)
        float mx[8], sm[8];
#pragma unroll
        for (int v = 0; v < 8; v++) {
            float m = -1e30f;
#pragma unroll
            for (int nt = 0; nt < 14; nt++) m = fmaxf(m, acc[nt][v]);
            mx[v] = m;
        }
#pragma unroll
        for (int d = 1; d < 16; d <<= 1)
#pragma unroll
            for (int v = 0; v < 8; v++)
                mx[v] = fmaxf(mx[v], __shfl_xor(mx[v], d, 32));
#pragma unroll
        for (int v = 0; v < 8; v++) {
            float s = 0.f;
#pragma unroll
            for (int nt = 0; nt < 14; nt++) {
                const float e = __expf(acc[nt][v] - mx[v]);
                acc[nt][v] = e;
                s += e;
            }
            sm[v] = s;
        }
#pragma unroll
        for (int d = 1; d < 16; d <<= 1)
#pragma unroll
            for (int v = 0; v < 8; v++) sm[v] += __shfl_xor(sm[v], d, 32);

        // ---- P (bf16) -> per-wave LDS (C layout -> A layout reshape)
        unsigned short* pw = &Pl[wid][0];
#pragma unroll
        for (int v = 0; v < 8; v++) {
            const float inv = 1.f / sm[v];
#pragma unroll
            for (int nt = 0; nt < 14; nt++)
                pw[(hi * 8 + v) * 224 + nt * 16 + l15] = f2bf(acc[nt][v] * inv);
        }

        // ---- O = P V   (K = 224 = 7 x 32)
#pragma unroll
        for (int dt = 0; dt < 4; dt++) {
            v8f o = (v8f){0.f, 0.f, 0.f, 0.f, 0.f, 0.f, 0.f, 0.f};
#pragma unroll
            for (int kk = 0; kk < 7; kk++) {
                Frag pf, vf;
                const unsigned short* pp = pw + l15 * 224 + kk * 32 + hi * 16;
                pf.q[0] = *(const uint4*)pp;
                pf.q[1] = *(const uint4*)(pp + 8);
                const unsigned short* vp =
                    &VT[(dt * 16 + l15) * LVT + kk * 32 + hi * 16];
                vf.q[0] = ((const uint4*)vp)[0];
                vf.q[1] = ((const uint4*)vp)[1];
                o = __builtin_amdgcn_wmma_f32_16x16x32_bf16(
                    false, pf.v, false, vf.v, (short)0, o, false, false);
            }
#pragma unroll
            for (int v = 0; v < 8; v++) {
                const int tq2 = mt * 16 + hi * 8 + v;
                if (tq2 < 196)
                    ab[((size_t)b * 196 + tq2) * 768 + h * 64 + dt * 16 + l15] =
                        f2bf(o[v]);
            }
        }
    }
}

// ---------------------------------------------------------------------------
extern "C" void kernel_launch(void* const* d_in, const int* in_sizes, int n_in,
                              void* d_out, int out_size, void* d_ws, size_t ws_size,
                              hipStream_t stream) {
    (void)in_sizes; (void)n_in; (void)out_size; (void)ws_size;
    const float* x      = (const float*)d_in[0];  // [128,196,768]
    const float* qkv_w  = (const float*)d_in[1];  // [2304,768]
    const float* q_bias = (const float*)d_in[2];  // [768]
    const float* v_bias = (const float*)d_in[3];  // [768]
    const float* rpb    = (const float*)d_in[4];  // [729,12]
    const float* proj_w = (const float*)d_in[5];  // [768,768]
    const float* proj_b = (const float*)d_in[6];  // [768]
    float* out = (float*)d_out;                   // [25088,768]

    char* w = (char*)d_ws;
    size_t off = 0;
    auto takeb = [&](size_t bytes) -> void* {
        void* p = w + off;
        off += ((bytes + 255) & ~(size_t)255);
        return p;
    };
    unsigned short* xb   = (unsigned short*)takeb(25088ull * 768 * 2);
    unsigned short* wqb  = (unsigned short*)takeb(2304ull * 768 * 2);
    unsigned short* pwb  = (unsigned short*)takeb(768ull * 768 * 2);
    unsigned short* qkvb = (unsigned short*)takeb(25088ull * 2304 * 2);
    unsigned short* abuf = (unsigned short*)takeb(25088ull * 768 * 2);
    float*          bpad = (float*)takeb(12ull * 208 * 224 * 4);

    const int NT = 256;
    auto cgrid = [](int n) { int g = (n + 255) / 256; return g > 32768 ? 32768 : g; };
    f32_to_bf16_kernel<<<cgrid(25088 * 768), NT, 0, stream>>>(x, xb, 25088 * 768);
    f32_to_bf16_kernel<<<cgrid(2304 * 768),  NT, 0, stream>>>(qkv_w, wqb, 2304 * 768);
    f32_to_bf16_kernel<<<cgrid(768 * 768),   NT, 0, stream>>>(proj_w, pwb, 768 * 768);
    build_bias_kernel<<<(12 * 208 * 224 + NT - 1) / NT, NT, 0, stream>>>(rpb, bpad);

    // QKV GEMM: [25088 x 768] x [768 x 2304]
    gemm_bf16_kernel<0><<<dim3(2304 / 128, 25088 / 128), 256, 0, stream>>>(
        xb, wqb, qkvb, nullptr, q_bias, v_bias, nullptr, 25088, 2304, 768);

    // Attention per (b, h)
    attn_kernel<<<128 * 12, 256, 0, stream>>>(qkvb, bpad, abuf);

    // Projection GEMM: [25088 x 768] x [768 x 768] -> fp32 out
    gemm_bf16_kernel<1><<<dim3(768 / 128, 25088 / 128), 256, 0, stream>>>(
        abuf, pwb, nullptr, out, nullptr, nullptr, proj_b, 25088, 768, 768);
}
